// SAE_77060303225533
// MI455X (gfx1250) — compile-verified, matrix operand order checked by reading, not run on hardware
//
#include <hip/hip_runtime.h>
#include <hip/hip_bf16.h>

// ---------------------------------------------------------------------------
// SAE TopK forward for MI455X (gfx1250, wave32, WMMA).
// Heavy GEMMs: v_wmma_f32_16x16x32_bf16 with B tiles staged into LDS via
// async global->LDS copies (double-buffered, s_wait_asynccnt synchronized).
// ---------------------------------------------------------------------------

#define Bt   2
#define Dd   4
#define Ll   512
#define Hh   1024
#define Ff   16384
#define Nn   (Bt * Dd * Ll)      // 4096 tokens
#define TOPK 64
#define AUXK 512
#define DEAD_THRESH 200000

typedef __attribute__((ext_vector_type(16))) __bf16 v16bf;
typedef __attribute__((ext_vector_type(8)))  float  v8f;

__device__ __forceinline__ unsigned short f32_to_bf16_rne(float f) {
    unsigned int u = __builtin_bit_cast(unsigned int, f);
    unsigned int r = u + 0x7FFFu + ((u >> 16) & 1u);
    return (unsigned short)(r >> 16);
}

__device__ __forceinline__ void wait_async_zero() {
#if __has_builtin(__builtin_amdgcn_s_wait_asynccnt)
    __builtin_amdgcn_s_wait_asynccnt(0);
#else
    asm volatile("s_wait_asynccnt 0x0" ::: "memory");
#endif
}

__device__ __forceinline__ float block_reduce_sum(float v) {
    for (int off = 16; off > 0; off >>= 1) v += __shfl_down(v, off, 32);
    __shared__ float s[32];
    const int lane = threadIdx.x & 31, w = threadIdx.x >> 5;
    if (lane == 0) s[w] = v;
    __syncthreads();
    const int nw = (blockDim.x + 31) >> 5;
    v = (threadIdx.x < (unsigned)nw) ? s[threadIdx.x] : 0.f;
    if (w == 0) for (int off = 16; off > 0; off >>= 1) v += __shfl_down(v, off, 32);
    return v;   // valid in thread 0
}

// ---------------------------------------------------------------------------
// init: zero accumulators + active flags
// ---------------------------------------------------------------------------
__global__ void k_init(int* active, float* accum) {
    int i = blockIdx.x * blockDim.x + threadIdx.x;
    if (i < Ff) active[i] = 0;
    if (i < 2)  accum[i] = 0.f;
}

// ---------------------------------------------------------------------------
// f32 -> bf16 conversion (4 elems / thread)
// ---------------------------------------------------------------------------
__global__ void k_f32_to_bf16(const float* __restrict__ src,
                              unsigned short* __restrict__ dst, int n) {
    int i = (blockIdx.x * blockDim.x + threadIdx.x) * 4;
    if (i + 3 < n) {
        const float4 v = *(const float4*)(src + i);
        dst[i + 0] = f32_to_bf16_rne(v.x);
        dst[i + 1] = f32_to_bf16_rne(v.y);
        dst[i + 2] = f32_to_bf16_rne(v.z);
        dst[i + 3] = f32_to_bf16_rne(v.w);
    }
}

// xb = bf16(x - bias_pre[h])   (flat over Nn*Hh, h = i & 1023)
__global__ void k_sub_bias_bf16(const float* __restrict__ x,
                                const float* __restrict__ bias_pre,
                                unsigned short* __restrict__ dst, int n) {
    int i = blockIdx.x * blockDim.x + threadIdx.x;
    if (i < n) dst[i] = f32_to_bf16_rne(x[i] - bias_pre[i & (Hh - 1)]);
}

// ---------------------------------------------------------------------------
// Generic bf16 WMMA GEMM: C[m][n] = sum_k A[m][k]*B[n][k]  (both K-contiguous)
// block = 256 threads = 8 waves; wave tile 16(M) x 64(N); block tile 128 x 64.
// B tile (64x32 bf16 = 4KB) staged into LDS with async global->LDS b128
// copies, double-buffered (2 x 4KB), shared by all 8 waves (8x traffic cut).
// Epilogue: +bias[n] (opt), relu (opt), *colScale[n] (opt).
// ---------------------------------------------------------------------------
union Frag16 { unsigned int u[8]; v16bf v; };

#define GEMM_KS 32   // K per step (bf16 WMMA depth)
#define GEMM_BN 64   // N per block

__global__ void k_gemm_bf16(const unsigned short* __restrict__ A, int lda,
                            const unsigned short* __restrict__ B, int ldb,
                            float* __restrict__ C, int ldc,
                            int M, int N, int K,
                            const float* __restrict__ bias,
                            const float* __restrict__ colScale,
                            int doRelu) {
    __shared__ __align__(16) unsigned short sB[2][GEMM_BN * GEMM_KS]; // 2 x 4KB
    const int tid  = threadIdx.x;
    const int lane = tid & 31;
    const int wave = tid >> 5;                       // 0..7
    const int m0   = blockIdx.y * 128 + wave * 16;
    const int n0   = blockIdx.x * GEMM_BN;
    const int r    = lane & 15;
    const int hi   = (lane >> 4) << 3;               // 0 for lanes 0-15, 8 for 16-31

    // Async staging assignment: 256 threads x 16B = 4KB tile per k-step.
    const int tRow   = tid >> 2;                     // 0..63 (tile row in N)
    const int tChunk = (tid & 3) * 8;                // element offset (8 bf16 = 16B)
    const unsigned short* gB = B + (size_t)(n0 + tRow) * ldb + tChunk;
    const unsigned ldsDst0 = (unsigned)(uintptr_t)&sB[0][tRow * GEMM_KS + tChunk];
    const unsigned ldsDst1 = (unsigned)(uintptr_t)&sB[1][tRow * GEMM_KS + tChunk];

#define ISSUE_ASYNC(ldsOff, kk)                                              \
    asm volatile("global_load_async_to_lds_b128 %0, %1, off"                 \
                 :: "v"(ldsOff),                                             \
                    "v"((unsigned long long)(uintptr_t)(gB + (kk)))          \
                 : "memory")

    v8f acc[4];
    #pragma unroll
    for (int t = 0; t < 4; ++t) acc[t] = (v8f){0.f,0.f,0.f,0.f,0.f,0.f,0.f,0.f};

    ISSUE_ASYNC(ldsDst0, 0);                         // prologue: fill buf0

    for (int k0 = 0; k0 < K; k0 += GEMM_KS) {
        const int cur = (k0 >> 5) & 1;
        // Our own async for `cur` is the only one outstanding -> wait it,
        // then barrier so every wave's staged portion is visible.
        wait_async_zero();
        __syncthreads();
        if (k0 + GEMM_KS < K) {                      // prefetch next tile
            if (cur == 0) { ISSUE_ASYNC(ldsDst1, k0 + GEMM_KS); }
            else          { ISSUE_ASYNC(ldsDst0, k0 + GEMM_KS); }
        }

        // A fragment: 16x32 bf16 direct from global, lane row = m0 + r.
        Frag16 fa;
        const unsigned short* arow = A + (size_t)(m0 + r) * lda + k0;
        #pragma unroll
        for (int i = 0; i < 8; ++i) {
            const int koff = (i < 4) ? (hi + 2 * i) : (16 + hi + 2 * (i - 4));
            fa.u[i] = *(const unsigned int*)(const void*)(arow + koff);
        }
        // B fragments from LDS (pair offsets coalesce to ds_load_b128).
        #pragma unroll
        for (int t = 0; t < 4; ++t) {
            Frag16 fb;
            const unsigned short* bls = &sB[cur][(t * 16 + r) * GEMM_KS];
            #pragma unroll
            for (int i = 0; i < 8; ++i) {
                const int koff = (i < 4) ? (hi + 2 * i) : (16 + hi + 2 * (i - 4));
                fb.u[i] = *(const unsigned int*)(const void*)(bls + koff);
            }
            acc[t] = __builtin_amdgcn_wmma_f32_16x16x32_bf16(
                         false, fa.v, false, fb.v, (short)0, acc[t], false, false);
        }
    }
#undef ISSUE_ASYNC

    // Epilogue. C layout: VGPR j -> (M = j + (lane<16?0:8), N = lane&15)
    const int mAdd = (lane >> 4) << 3;
    #pragma unroll
    for (int t = 0; t < 4; ++t) {
        const int n  = n0 + t * 16 + r;
        const float bs = bias     ? bias[n]     : 0.f;
        const float cs = colScale ? colScale[n] : 1.f;
        #pragma unroll
        for (int j = 0; j < 8; ++j) {
            const int m = m0 + j + mAdd;
            float v = acc[t][j] + bs;
            if (doRelu) v = v > 0.f ? v : 0.f;
            v *= cs;
            C[(size_t)m * ldc + n] = v;
        }
    }
}

// ---------------------------------------------------------------------------
// Per-row TopK-64 over 16384 (row cached in exactly 64KB LDS).
// Packed key: positive-f32 bits are order-monotonic; low word breaks ties
// toward the lowest index (matches jax.lax.top_k).
// ---------------------------------------------------------------------------
__global__ void k_topk(const float* __restrict__ zdense,
                       float* __restrict__ topVals, int* __restrict__ topIdx,
                       unsigned long long* __restrict__ redScratch) {
    const int row = blockIdx.x;
    const float* zrow = zdense + (size_t)row * Ff;
    __shared__ float sv[Ff];                               // 64 KB
    for (int i = threadIdx.x; i < Ff; i += blockDim.x) sv[i] = zrow[i];
    __syncthreads();

    unsigned long long* red = redScratch + (size_t)row * 16;
    const int lane = threadIdx.x & 31, wv = threadIdx.x >> 5;
    const int nw = blockDim.x >> 5;

    for (int t = 0; t < TOPK; ++t) {
        unsigned long long best = 0ULL;
        for (int i = threadIdx.x; i < Ff; i += blockDim.x) {
            const float v = sv[i];
            unsigned long long key = (v < 0.f) ? 0ULL
                : (((unsigned long long)__builtin_bit_cast(unsigned int, v)) << 32)
                  | (0xFFFFFFFFu - (unsigned)i);
            if (key > best) best = key;
        }
        for (int off = 16; off > 0; off >>= 1) {
            unsigned long long o = __shfl_down(best, off, 32);
            if (o > best) best = o;
        }
        if (lane == 0) red[wv] = best;
        __threadfence_block();
        __syncthreads();
        if (threadIdx.x == 0) {
            unsigned long long b = 0ULL;
            for (int w2 = 0; w2 < nw; ++w2) if (red[w2] > b) b = red[w2];
            const int   idx = (int)(0xFFFFFFFFu - (unsigned)(b & 0xFFFFFFFFu));
            const float val = __builtin_bit_cast(float, (unsigned)(b >> 32));
            topVals[row * TOPK + t] = val;
            topIdx[row * TOPK + t]  = idx;
            sv[idx] = -1.f;                                // mark taken
        }
        __syncthreads();
    }
}

// ---------------------------------------------------------------------------
// Sparse decode: x_tgt = z_sparse @ W + bias_pre ; e = x - x_tgt ;
// recon partial sums ; active-feature marking.
// ---------------------------------------------------------------------------
__global__ void k_decode(const float* __restrict__ topVals,
                         const int* __restrict__ topIdx,
                         const float* __restrict__ W,        // dictionary_enc, Ff x Hh
                         const float* __restrict__ bias_pre,
                         const float* __restrict__ x,        // zL flat, Nn x Hh
                         const unsigned char* __restrict__ mask,
                         float* __restrict__ xTgtOut,        // d_out + 3
                         float* __restrict__ eOut,
                         int* __restrict__ active,
                         float* __restrict__ reconAccum) {
    const int row = blockIdx.x;
    __shared__ float svv[TOPK];
    __shared__ int   sii[TOPK];
    if (threadIdx.x < TOPK) {
        const float v = topVals[row * TOPK + threadIdx.x];
        const int   i = topIdx[row * TOPK + threadIdx.x];
        svv[threadIdx.x] = v;
        sii[threadIdx.x] = i;
        if (v > 0.f) active[i] = 1;                        // benign races
    }
    __syncthreads();

    const int h0 = threadIdx.x * 4;
    float a0 = 0.f, a1 = 0.f, a2 = 0.f, a3 = 0.f;
    #pragma unroll 8
    for (int k = 0; k < TOPK; ++k) {
        const float  z = svv[k];
        const float4 w = *(const float4*)(W + (size_t)sii[k] * Hh + h0);
        a0 += z * w.x; a1 += z * w.y; a2 += z * w.z; a3 += z * w.w;
    }
    const float4 bp = *(const float4*)(bias_pre + h0);
    a0 += bp.x; a1 += bp.y; a2 += bp.z; a3 += bp.w;

    const size_t base = (size_t)row * Hh + h0;
    xTgtOut[base + 0] = a0; xTgtOut[base + 1] = a1;
    xTgtOut[base + 2] = a2; xTgtOut[base + 3] = a3;

    const float4 xv = *(const float4*)(x + base);
    float4 e; e.x = xv.x - a0; e.y = xv.y - a1; e.z = xv.z - a2; e.w = xv.w - a3;
    *(float4*)(eOut + base) = e;

    const int b = row / (Dd * Ll), l = row & (Ll - 1);
    const float mf = mask[b * Ll + l] ? 1.f : 0.f;
    float sq = (e.x * e.x + e.y * e.y + e.z * e.z + e.w * e.w) * mf;
    sq = block_reduce_sum(sq);
    if (threadIdx.x == 0) atomicAdd(reconAccum, sq);
}

// ---------------------------------------------------------------------------
// Dead-feature top-512 (single block, 1024 threads, 64KB LDS of scores).
// ---------------------------------------------------------------------------
__global__ void k_dead_select(const int* __restrict__ lastActive,
                              const int* __restrict__ numSeen,
                              const int* __restrict__ active,
                              int* __restrict__ selected,
                              float* __restrict__ validArr,
                              unsigned long long* __restrict__ red) {
    __shared__ int sc[Ff];                                 // 64 KB
    const int newSeen = numSeen[0] + Nn;
    for (int i = threadIdx.x; i < Ff; i += blockDim.x) {
        const int age = active[i] ? 0 : (newSeen - lastActive[i]);
        sc[i] = (age >= DEAD_THRESH) ? age : -1;
    }
    __syncthreads();

    const int lane = threadIdx.x & 31, wv = threadIdx.x >> 5;
    const int nw = blockDim.x >> 5;
    for (int t = 0; t < AUXK; ++t) {
        unsigned long long best = 0ULL;
        for (int i = threadIdx.x; i < Ff; i += blockDim.x) {
            const int s = sc[i];
            if (s == INT_MIN) continue;                    // already taken
            const unsigned long long key =
                (((unsigned long long)((unsigned)s ^ 0x80000000u)) << 32)
                | (0xFFFFFFFFu - (unsigned)i);
            if (key > best) best = key;
        }
        for (int off = 16; off > 0; off >>= 1) {
            unsigned long long o = __shfl_down(best, off, 32);
            if (o > best) best = o;
        }
        if (lane == 0) red[wv] = best;
        __threadfence_block();
        __syncthreads();
        if (threadIdx.x == 0) {
            unsigned long long b = 0ULL;
            for (int w2 = 0; w2 < nw; ++w2) if (red[w2] > b) b = red[w2];
            const int idx   = (int)(0xFFFFFFFFu - (unsigned)(b & 0xFFFFFFFFu));
            const int score = (int)((unsigned)(b >> 32) ^ 0x80000000u);
            selected[t] = idx;
            validArr[t] = (score >= DEAD_THRESH) ? 1.f : 0.f;
            sc[idx] = INT_MIN;
        }
        __syncthreads();
    }
}

// Gather dead rows of bf16 W (plus transposed copy for the decode GEMM) + bias.
__global__ void k_gather_dead(const unsigned short* __restrict__ Wbf,
                              const float* __restrict__ biasEnc,
                              const int* __restrict__ selected,
                              unsigned short* __restrict__ Wd,    // AUXK x Hh
                              unsigned short* __restrict__ WdT,   // Hh x AUXK
                              float* __restrict__ bDead) {
    const int j = blockIdx.x;
    const int sel = selected[j];
    for (int h = threadIdx.x; h < Hh; h += blockDim.x) {
        const unsigned short w = Wbf[(size_t)sel * Hh + h];
        Wd [(size_t)j * Hh + h]  = w;
        WdT[(size_t)h * AUXK + j] = w;
    }
    if (threadIdx.x == 0) bDead[j] = biasEnc[sel];
}

// aux squared error: sum((ehat - e)^2 * mask)
__global__ void k_aux_sq(const float* __restrict__ ehat,
                         const float* __restrict__ e,
                         const unsigned char* __restrict__ mask,
                         float* __restrict__ auxAccum) {
    const int i0 = (blockIdx.x * blockDim.x + threadIdx.x) * 4;
    const int row = i0 >> 10;                              // /Hh
    const int b = row / (Dd * Ll), l = row & (Ll - 1);
    const float mf = mask[b * Ll + l] ? 1.f : 0.f;
    const float4 a = *(const float4*)(ehat + i0);
    const float4 c = *(const float4*)(e + i0);
    const float dx = a.x - c.x, dy = a.y - c.y, dz = a.z - c.z, dw = a.w - c.w;
    float sq = (dx * dx + dy * dy + dz * dz + dw * dw) * mf;
    sq = block_reduce_sum(sq);
    if (threadIdx.x == 0) atomicAdd(auxAccum, sq);
}

// finalize: losses
__global__ void k_finalize(const unsigned char* __restrict__ mask,
                           const float* __restrict__ accum,
                           float* __restrict__ out) {
    float s = 0.f;
    for (int i = threadIdx.x; i < Bt * Ll; i += blockDim.x) s += mask[i] ? 1.f : 0.f;
    s = block_reduce_sum(s);
    if (threadIdx.x == 0) {
        float nv = s * (float)Dd * (float)Hh;
        nv = nv > 1.f ? nv : 1.f;
        const float recon = accum[0] / nv;
        const float aux   = accum[1] / nv;
        out[0] = recon + aux * (1.f / 32.f);
        out[1] = recon;
        out[2] = aux;
    }
}

// ---------------------------------------------------------------------------
// Host-side launch. Workspace layout (needs ~371 MB; dominated by dense z).
// ---------------------------------------------------------------------------
extern "C" void kernel_launch(void* const* d_in, const int* in_sizes, int n_in,
                              void* d_out, int out_size, void* d_ws, size_t ws_size,
                              hipStream_t stream) {
    const float*         zL       = (const float*)d_in[0];          // Nn x Hh
    const unsigned char* mask     = (const unsigned char*)d_in[1];  // Bt x Ll
    const float*         Wenc     = (const float*)d_in[2];          // Ff x Hh
    const float*         biasPre  = (const float*)d_in[4];          // Hh
    const float*         biasEnc  = (const float*)d_in[5];          // Ff
    const int*           lastAct  = (const int*)d_in[6];            // Ff
    const int*           numSeen  = (const int*)d_in[7];            // scalar

    float* out   = (float*)d_out;          // [loss, recon, aux, x_tgt...]
    float* xTgt  = out + 3;

    char* w = (char*)d_ws;
    size_t off = 0;
    auto carve = [&](size_t bytes) { void* p = w + off; off = (off + bytes + 255) & ~(size_t)255; return p; };
    unsigned short* Wbf   = (unsigned short*)carve((size_t)Ff * Hh * 2);   // 32 MB
    unsigned short* Xb    = (unsigned short*)carve((size_t)Nn * Hh * 2);   //  8 MB
    float*          Zd    = (float*)carve((size_t)Nn * Ff * 4);            // 256 MB
    float*          topV  = (float*)carve((size_t)Nn * TOPK * 4);
    int*            topI  = (int*)carve((size_t)Nn * TOPK * 4);
    int*            act   = (int*)carve((size_t)Ff * 4);
    float*          eBuf  = (float*)carve((size_t)Nn * Hh * 4);            // 16 MB
    unsigned short* Eb    = (unsigned short*)carve((size_t)Nn * Hh * 2);
    unsigned short* Wd    = (unsigned short*)carve((size_t)AUXK * Hh * 2);
    unsigned short* WdT   = (unsigned short*)carve((size_t)Hh * AUXK * 2);
    float*          bDead = (float*)carve(AUXK * 4);
    float*          valid = (float*)carve(AUXK * 4);
    int*            sel   = (int*)carve(AUXK * 4);
    float*          zAux  = (float*)carve((size_t)Nn * AUXK * 4);          //  8 MB
    unsigned short* zAuxB = (unsigned short*)carve((size_t)Nn * AUXK * 2);
    float*          eHat  = (float*)carve((size_t)Nn * Hh * 4);            // 16 MB
    float*          accum = (float*)carve(256);
    unsigned long long* redTop  = (unsigned long long*)carve((size_t)Nn * 16 * 8);
    unsigned long long* redDead = (unsigned long long*)carve(32 * 8);
    (void)ws_size; (void)in_sizes; (void)n_in; (void)out_size;

    // 1) init accumulators / active flags
    k_init<<<(Ff + 255) / 256, 256, 0, stream>>>(act, accum);
    // 2) W f32 -> bf16 (reused by main + aux paths)
    k_f32_to_bf16<<<((size_t)Ff * Hh / 4 + 255) / 256, 256, 0, stream>>>(Wenc, Wbf, Ff * Hh);
    // 3) Xb = bf16(x - bias_pre)
    k_sub_bias_bf16<<<((size_t)Nn * Hh + 255) / 256, 256, 0, stream>>>(zL, biasPre, Xb, Nn * Hh);
    // 4) encoder GEMM: z_dense = relu(Xb @ Wbf^T + bias_enc)   (WMMA bf16)
    k_gemm_bf16<<<dim3(Ff / 64, Nn / 128), 256, 0, stream>>>(
        Xb, Hh, Wbf, Hh, Zd, Ff, Nn, Ff, Hh, biasEnc, nullptr, 1);
    // 5) per-row TopK-64
    k_topk<<<Nn, 512, 0, stream>>>(Zd, topV, topI, redTop);
    // 6) sparse decode + e + recon sum + active flags
    k_decode<<<Nn, 256, 0, stream>>>(topV, topI, Wenc, biasPre, zL, mask,
                                     xTgt, eBuf, act, accum + 0);
    // 7) dead-feature top-512
    k_dead_select<<<1, 1024, 0, stream>>>(lastAct, numSeen, act, sel, valid, redDead);
    // 8) gather dead dictionary rows (bf16) + bias
    k_gather_dead<<<AUXK, 256, 0, stream>>>(Wbf, biasEnc, sel, Wd, WdT, bDead);
    // 9) Eb = bf16(e - bias_pre)
    k_sub_bias_bf16<<<((size_t)Nn * Hh + 255) / 256, 256, 0, stream>>>(eBuf, biasPre, Eb, Nn * Hh);
    // 10) aux encoder GEMM: z_aux = relu(Eb @ Wd^T + bDead) * valid   (WMMA)
    k_gemm_bf16<<<dim3(AUXK / 64, Nn / 128), 256, 0, stream>>>(
        Eb, Hh, Wd, Hh, zAux, AUXK, Nn, AUXK, Hh, bDead, valid, 1);
    // 11) z_aux -> bf16
    k_f32_to_bf16<<<((size_t)Nn * AUXK / 4 + 255) / 256, 256, 0, stream>>>(zAux, zAuxB, Nn * AUXK);
    // 12) aux decoder GEMM: e_hat = zAuxB @ WdT^T   (WMMA)
    k_gemm_bf16<<<dim3(Hh / 64, Nn / 128), 256, 0, stream>>>(
        zAuxB, AUXK, WdT, AUXK, eHat, Hh, Nn, Hh, AUXK, nullptr, nullptr, 0);
    // 13) aux squared-error sum
    k_aux_sq<<<(size_t)Nn * Hh / (256 * 4), 256, 0, stream>>>(eHat, eBuf, mask, accum + 1);
    // 14) finalize losses
    k_finalize<<<1, 256, 0, stream>>>(mask, accum, out);
}
